// GroupedExperts_32701880992122
// MI455X (gfx1250) — compile-verified, matrix operand order checked by reading, not run on hardware
//
#include <hip/hip_runtime.h>

#define NE   8
#define DIM  1024
#define HID  2048
#define NT   8192
#define MT   128         // token tile (M) per workgroup
#define HC   128         // hidden chunk
#define XSTR (DIM + 8)   // LDS row stride for x tile (bank stagger, 16B aligned)
#define GSTR (HC + 8)    // LDS row stride for g tile
#define MAX_TILES (NT / MT)  // 64: covers up to 8192 tokens in one expert
#define NTHREADS 1024    // 32 waves

typedef __attribute__((ext_vector_type(16))) __bf16 v16bf;
typedef __attribute__((ext_vector_type(8)))  __bf16 v8bf;
typedef __attribute__((ext_vector_type(4)))  __bf16 v4bf;
typedef __attribute__((ext_vector_type(8)))  float  v8f;
typedef __attribute__((ext_vector_type(4)))  float  v4f;

union V16 { v16bf v; v8bf h[2]; };

__device__ __forceinline__ v16bf cat8(v8bf lo, v8bf hi) {
  V16 t; t.h[0] = lo; t.h[1] = hi; return t.v;
}

// Pack 16 f32 (k..k+7 and k+16..k+23) into a bf16 WMMA operand fragment.
__device__ __forceinline__ v16bf cvt16(v4f a, v4f b, v4f c, v4f d) {
  V16 t;
#pragma unroll
  for (int i = 0; i < 4; ++i) {
    t.v[i]      = (__bf16)a[i];
    t.v[4 + i]  = (__bf16)b[i];
    t.v[8 + i]  = (__bf16)c[i];
    t.v[12 + i] = (__bf16)d[i];
  }
  return t.v;
}

// B-matrix fragment: lane holds column n = lane%16 of B (= one row of W),
// K elements {kA..kA+7, kA+16..kA+23}; rows of W are contiguous in K.
__device__ __forceinline__ v16bf load_b_global(const float* __restrict__ row, int kA) {
  v4f a = *(const v4f*)(row + kA);
  v4f b = *(const v4f*)(row + kA + 4);
  v4f c = *(const v4f*)(row + kA + 16);
  v4f d = *(const v4f*)(row + kA + 20);
  return cvt16(a, b, c, d);
}

__device__ __forceinline__ v8f wmma_bf16(v16bf A, v16bf B, v8f C) {
  return __builtin_amdgcn_wmma_f32_16x16x32_bf16(
      /*neg_a=*/false, A, /*neg_b=*/false, B,
      /*c_mod=*/(short)0, C, /*reuse_a=*/false, /*reuse_b=*/false);
}

// Zero only the padding tail (rows >= sum(counts)); valid rows exit instantly.
__global__ __launch_bounds__(256) void moe_zero_tail_kernel(
    float* __restrict__ out, const long long* __restrict__ cnts) {
  long long total = 0;
  for (int i = 0; i < NE; ++i) total += cnts[i];
  const int r = blockIdx.x;
  if ((long long)r < total) return;
  v4f z = {0.f, 0.f, 0.f, 0.f};
  *(v4f*)(out + (size_t)r * DIM + threadIdx.x * 4) = z;
}

__global__ __launch_bounds__(NTHREADS) void moe_ffn_kernel(
    const float* __restrict__ x, const long long* __restrict__ cnts,
    const float* __restrict__ w1, const float* __restrict__ w2,
    const float* __restrict__ w3, float* __restrict__ out) {
  __shared__ __bf16 xs[MT * XSTR];   // 128 x 1024 x-tile (bf16): 264 KB
  __shared__ __bf16 gs[MT * GSTR];   // 128 x 128 g-tile (bf16): 35 KB

  const int e    = blockIdx.x / MAX_TILES;
  const int tile = blockIdx.x % MAX_TILES;

  long long off = 0;
  for (int i = 0; i < e; ++i) off += cnts[i];
  const long long ce = cnts[e];
  const long long t0 = (long long)tile * MT;
  if (t0 >= ce) return;                       // uniform early exit
  const int row0 = (int)(off + t0);
  const int mv   = (int)((ce - t0 < (long long)MT) ? (ce - t0) : (long long)MT);

  const int tid   = threadIdx.x;
  const int wave  = tid >> 5;                 // 0..31
  const int lane  = tid & 31;
  const int lhalf = lane >> 4;                // 0: K 0..7/16..23, 1: K 8..15/24..31
  const int lmod  = lane & 15;

  // ---- Stage x tile into LDS as bf16; rows beyond the expert slice are zero.
#pragma unroll
  for (int it = 0; it < (MT * DIM) / (4 * NTHREADS); ++it) {   // 32 iters
    const int q = it * NTHREADS + tid;        // quad index
    const int r = q >> 8;                     // 256 quads per row
    const int c = (q & 255) * 4;
    v4f v = {0.f, 0.f, 0.f, 0.f};
    if (r < mv) v = *(const v4f*)(x + (size_t)(row0 + r) * DIM + c);
    v4bf b;
    b[0] = (__bf16)v[0]; b[1] = (__bf16)v[1];
    b[2] = (__bf16)v[2]; b[3] = (__bf16)v[3];
    *(v4bf*)(xs + r * XSTR + c) = b;
  }
  __syncthreads();

  // ---- Persistent f32 accumulators: wave owns all 128 tokens x 32 dims.
  v8f zero8 = {0.f, 0.f, 0.f, 0.f, 0.f, 0.f, 0.f, 0.f};
  v8f acc[8][2];                              // [m block][n tile]: 128 VGPRs
#pragma unroll
  for (int i = 0; i < 8; ++i)
#pragma unroll
    for (int j = 0; j < 2; ++j) acc[i][j] = zero8;

  const float* w1e = w1 + (size_t)e * HID * DIM;
  const float* w3e = w3 + (size_t)e * HID * DIM;
  const float* w2e = w2 + (size_t)e * DIM * HID;
  const int dw = wave * 32;                   // this wave's output-dim base

  // Per-lane w2 row pointers (lane's B columns), hoisted out of the h-loop.
  const float* w2r0 = w2e + (size_t)(dw + lmod) * HID;
  const float* w2r1 = w2e + (size_t)(dw + 16 + lmod) * HID;

  const int hq = wave >> 2;                   // phase-A hidden tile 0..7
  const int mp = (wave & 3) * 2;              // phase-A m blocks {mp, mp+1}

  for (int h0 = 0; h0 < HID; h0 += HC) {
    // ===== Phase A: g = silu(x@w1^T) * (x@w3^T), 128 x HC ================
    // wave computes hidden cols [h0+16*hq, +16) for m blocks mp, mp+1.
    const float* r1 = w1e + (size_t)(h0 + hq * 16 + lmod) * DIM;
    const float* r3 = w3e + (size_t)(h0 + hq * 16 + lmod) * DIM;
    v8f a1[2] = {zero8, zero8}, a3[2] = {zero8, zero8};
    for (int k0 = 0; k0 < DIM; k0 += 32) {
      const int kA = k0 + lhalf * 8;
      __builtin_prefetch(r1 + kA + 128, 0, 1);   // global_prefetch next lines
      __builtin_prefetch(r3 + kA + 128, 0, 1);
      v16bf B1 = load_b_global(r1, kA);          // loaded once,
      v16bf B3 = load_b_global(r3, kA);          // reused for 2 m blocks
#pragma unroll
      for (int mi = 0; mi < 2; ++mi) {
        const int mr = 16 * (mp + mi) + lmod;
        v16bf A = cat8(*(const v8bf*)(xs + mr * XSTR + kA),
                       *(const v8bf*)(xs + mr * XSTR + kA + 16));
        a1[mi] = wmma_bf16(A, B1, a1[mi]);
        a3[mi] = wmma_bf16(A, B3, a3[mi]);
      }
    }

    __syncthreads();                          // prior phase-B reads of gs done
#pragma unroll
    for (int mi = 0; mi < 2; ++mi)
#pragma unroll
      for (int v = 0; v < 8; ++v) {
        const float p = a1[mi][v];
        // silu(p) = p * sigmoid(p); fast rcp is plenty at bf16 precision
        const float g = p * __builtin_amdgcn_rcpf(1.f + __expf(-p)) * a3[mi][v];
        const int m = 16 * (mp + mi) + v + 8 * lhalf;
        gs[m * GSTR + hq * 16 + lmod] = (__bf16)g;
      }
    __syncthreads();                          // gs complete for all waves

    // ===== Phase B: out += g @ w2[:, h0:h0+HC]^T ==========================
    // B fragments (16 VGPRs) loaded first and reused 8x; A fragment is
    // transient (8 VGPRs) -> peak pressure stays under the spill threshold.
#pragma unroll
    for (int kk = 0; kk < HC; kk += 32) {
      const int kA = kk + lhalf * 8;
      v16bf B20 = load_b_global(w2r0 + h0, kA);
      v16bf B21 = load_b_global(w2r1 + h0, kA);
#pragma unroll
      for (int mi = 0; mi < 8; ++mi) {
        const int mr = 16 * mi + lmod;
        v16bf Ag = cat8(*(const v8bf*)(gs + mr * GSTR + kA),
                        *(const v8bf*)(gs + mr * GSTR + kA + 16));
        acc[mi][0] = wmma_bf16(Ag, B20, acc[mi][0]);
        acc[mi][1] = wmma_bf16(Ag, B21, acc[mi][1]);
      }
    }
  }

  // ---- Store: C/D layout VGPR v -> M = v + 8*(lane/16), N = lane%16.
#pragma unroll
  for (int mi = 0; mi < 8; ++mi)
#pragma unroll
    for (int nt = 0; nt < 2; ++nt)
#pragma unroll
      for (int v = 0; v < 8; ++v) {
        const int m = 16 * mi + v + 8 * lhalf;
        if (m < mv)
          out[(size_t)(row0 + m) * DIM + dw + nt * 16 + lmod] = acc[mi][nt][v];
      }
}

extern "C" void kernel_launch(void* const* d_in, const int* in_sizes, int n_in,
                              void* d_out, int out_size, void* d_ws, size_t ws_size,
                              hipStream_t stream) {
  const float*     x    = (const float*)d_in[0];
  const long long* cnts = (const long long*)d_in[1];
  const float*     w1   = (const float*)d_in[2];
  const float*     w2   = (const float*)d_in[3];
  const float*     w3   = (const float*)d_in[4];
  float* out = (float*)d_out;

  // Zero only padding rows (tokens beyond sum(counts)).
  moe_zero_tail_kernel<<<NT, 256, 0, stream>>>(out, cnts);
  moe_ffn_kernel<<<NE * MAX_TILES, NTHREADS, 0, stream>>>(x, cnts, w1, w2, w3, out);
}
